// MultiHeadAttentionBlock_25692494365196
// MI455X (gfx1250) — matrix operand order repaired and checked by reading, not compiled, not run on hardware
//
#include <hip/hip_runtime.h>
#include <hip/hip_bf16.h>

// ---- CDNA5 vector types --------------------------------------------------
typedef _Float16 v4h  __attribute__((ext_vector_type(4)));
typedef _Float16 v8h  __attribute__((ext_vector_type(8)));
typedef _Float16 v16h __attribute__((ext_vector_type(16)));
typedef float    v8f  __attribute__((ext_vector_type(8)));
typedef unsigned int u32x4 __attribute__((ext_vector_type(4)));
typedef int          i32x4 __attribute__((ext_vector_type(4)));
typedef int          i32x8 __attribute__((ext_vector_type(8)));

#if defined(__has_builtin)
#if __has_builtin(__builtin_amdgcn_tensor_load_to_lds)
#define USE_TDM 1
#else
#define USE_TDM 0
#endif
#else
#define USE_TDM 0
#endif

static __device__ __forceinline__ v16h cat16(v8h lo, v8h hi) {
  return __builtin_shufflevector(lo, hi, 0,1,2,3,4,5,6,7,8,9,10,11,12,13,14,15);
}

// A-matrix fragment (16x32 f16): lane l holds row M=l&15; half=l>>4.
// elems 0..15 -> K = 16*half + e (16 contiguous halves)
static __device__ __forceinline__ v16h load_a_frag(const _Float16* row_base_plus_khalf) {
  v8h lo = *(const v8h*)(row_base_plus_khalf);
  v8h hi = *(const v8h*)(row_base_plus_khalf + 8);
  return cat16(lo, hi);
}
// B-matrix fragment (32x16 f16): lane l holds col N=l&15; half=l>>4.
// elems 0..15 -> K = 16*half + e (16 contiguous halves)
static __device__ __forceinline__ v16h load_b_frag(const _Float16* col_base_plus_khalf) {
  v8h lo = *(const v8h*)(col_base_plus_khalf);
  v8h hi = *(const v8h*)(col_base_plus_khalf + 8);
  return cat16(lo, hi);
}

static __device__ __forceinline__ float half_red_max(float v) {
  for (int off = 1; off < 16; off <<= 1) v = fmaxf(v, __shfl_xor(v, off, 32));
  return v;
}
static __device__ __forceinline__ float half_red_sum(float v) {
  for (int off = 1; off < 16; off <<= 1) v += __shfl_xor(v, off, 32);
  return v;
}

#if USE_TDM
// Issue one Tensor-Data-Mover DMA: 2D f16 tile (tile_d0 x tile_d1 elements,
// row stride stride_d0 elements) from global -> LDS at lds_off.
// D# group0: count=1 | lds_addr | global_addr[56:0] | type=2.
// D# group1: data_size=2B, tensor dims == tile dims (tile fully in-bounds),
//            dim0 stride; dims >=2 unused (tile_dim2=0).
static __device__ __forceinline__ void tdm_load_tile_f16(
    unsigned int lds_off, const _Float16* gsrc,
    unsigned int tile_d0, unsigned int tile_d1, unsigned int stride_d0)
{
  const unsigned long long ga = (unsigned long long)(uintptr_t)gsrc;
  u32x4 g0;
  g0[0] = 1u;                                   // count=1, user mode
  g0[1] = lds_off;                              // lds_addr (bits 63:32)
  g0[2] = (unsigned int)ga;                     // global_addr[31:0]
  g0[3] = (unsigned int)((ga >> 32) & 0x01FFFFFFu) | 0x80000000u; // addr[56:32]|type=2
  i32x8 g1;
  g1[0] = 0x00010000;                           // workgroup_mask=0, data_size=2B
  g1[1] = (int)(tile_d0 << 16);                 // tensor_dim0[15:0] (bits 79:48 low)
  g1[2] = (int)(tile_d1 << 16);                 // tensor_dim1[15:0]
  g1[3] = (int)(tile_d0 << 16);                 // tile_dim0
  g1[4] = (int)tile_d1;                         // tile_dim1 (tile_dim2 = 0)
  g1[5] = (int)stride_d0;                       // tensor_dim0_stride[31:0]
  g1[6] = 0;
  g1[7] = 0;
  i32x4 z4 = {0, 0, 0, 0};
#if __has_include(<hip/amd_detail/amd_gfx1250_TDM.h>)
  i32x8 z8 = {0, 0, 0, 0, 0, 0, 0, 0};
  __builtin_amdgcn_tensor_load_to_lds(g0, g1, z4, z4, z8, 0);
#else
  __builtin_amdgcn_tensor_load_to_lds(g0, g1, z4, z4, 0);
#endif
}
#endif  // USE_TDM

// ==========================================================================
// GEMM: Y[4096,1024] = X[4096,1024] @ W[1024,1024] + bias   (f32 accumulate)
// MODE 0: X = f32, Out = f16 head-split [B,H,S,DK], val = (acc+bias)*scale
// MODE 1: X = f16 (TDM-staged A tiles), Out = f32 row-major [4096,1024]
// block = 256 threads (8 waves), tile 128x128, waves 4(M) x 2(N),
// each wave 32x64 = 2x4 WMMA tiles, K chunks of 32.
// ==========================================================================
template<int MODE>
__global__ __launch_bounds__(256)
void gemm_kernel(const void* __restrict__ Xv, const float* __restrict__ W,
                 const float* __restrict__ bias, void* __restrict__ Outv,
                 float scale)
{
  constexpr int Kd = 1024;
  const int tid  = threadIdx.x;
  const int lane = tid & 31;
  const int wave = tid >> 5;
  const int wm   = wave >> 1;       // 0..3
  const int wn   = wave & 1;        // 0..1
  const int half = lane >> 4;
  const int l15  = lane & 15;
  const int blockM = blockIdx.y * 128;
  const int blockN = blockIdx.x * 128;

  __shared__ alignas(16) _Float16 As[128 * 32];   // [m][k]
  __shared__ alignas(16) _Float16 Bt[128 * 32];   // [n][k] (transposed W tile)

  v8f zero = {};
  v8f acc[2][4];
  for (int i = 0; i < 2; ++i)
    for (int j = 0; j < 4; ++j) acc[i][j] = zero;

  const float*    Xf = (const float*)Xv;
  const _Float16* Xh = (const _Float16*)Xv;

  for (int k0 = 0; k0 < Kd; k0 += 32) {
    __syncthreads();
    // ---- stage A tile
    if (MODE == 0) {
      // f32 -> f16 conversion on the fly (TDM cannot convert)
      int r  = tid >> 3;          // 0..31
      int kq = (tid & 7) * 4;     // 0..28
      for (int rr = r; rr < 128; rr += 32) {
        const int grow = blockM + rr;
        const float4 f = *(const float4*)(Xf + (size_t)grow * Kd + k0 + kq);
        _Float16* d = As + rr * 32 + kq;
        d[0] = (_Float16)f.x; d[1] = (_Float16)f.y;
        d[2] = (_Float16)f.z; d[3] = (_Float16)f.w;
      }
    } else {
#if USE_TDM
      if (wave == 0)
        tdm_load_tile_f16((unsigned int)(uintptr_t)(void*)As,
                          Xh + (size_t)blockM * Kd + k0, 32u, 128u, (unsigned int)Kd);
#else
      int r  = tid >> 3;
      int kq = (tid & 7) * 4;
      for (int rr = r; rr < 128; rr += 32) {
        const int grow = blockM + rr;
        *(v4h*)(As + rr * 32 + kq) = *(const v4h*)(Xh + (size_t)grow * Kd + k0 + kq);
      }
#endif
    }
    // ---- stage W tile transposed: Bt[n][k] = W[k0+k][blockN+n]  (f32->f16)
    {
      int kk = tid >> 5;          // 0..7
      int n4 = (tid & 31) * 4;    // 0..124
      for (int k2 = kk; k2 < 32; k2 += 8) {
        const float4 w4 = *(const float4*)(W + (size_t)(k0 + k2) * 1024 + blockN + n4);
        Bt[(n4 + 0) * 32 + k2] = (_Float16)w4.x;
        Bt[(n4 + 1) * 32 + k2] = (_Float16)w4.y;
        Bt[(n4 + 2) * 32 + k2] = (_Float16)w4.z;
        Bt[(n4 + 3) * 32 + k2] = (_Float16)w4.w;
      }
    }
#if USE_TDM
    if (MODE == 1 && wave == 0)
      __builtin_amdgcn_s_wait_tensorcnt(0);
#endif
    __syncthreads();

    v16h a[2];
    for (int mt = 0; mt < 2; ++mt)
      a[mt] = load_a_frag(As + (wm * 32 + mt * 16 + l15) * 32 + 16 * half);
    for (int nt = 0; nt < 4; ++nt) {
      v16h b = load_b_frag(Bt + (wn * 64 + nt * 16 + l15) * 32 + 16 * half);
      for (int mt = 0; mt < 2; ++mt)
        acc[mt][nt] = __builtin_amdgcn_wmma_f32_16x16x32_f16(
            false, a[mt], false, b, (short)0, acc[mt][nt], false, false);
    }
  }

  // epilogue: C/D layout: VGPR r -> row r + 8*half, lane -> col l15
  _Float16* Oh = (_Float16*)Outv;
  float*    Of = (float*)Outv;
  for (int mt = 0; mt < 2; ++mt) {
    const int row0 = blockM + wm * 32 + mt * 16 + half * 8;
    for (int nt = 0; nt < 4; ++nt) {
      const int n  = blockN + wn * 64 + nt * 16 + l15;
      const float bn = bias[n];
      for (int r = 0; r < 8; ++r) {
        const int m = row0 + r;
        const float val = (acc[mt][nt][r] + bn) * scale;
        if (MODE == 0) {
          const int bb = m >> 11, s = m & 2047;
          const int hh = n >> 6,  dk = n & 63;
          Oh[(((size_t)bb * 16 + hh) * 2048 + s) * 64 + dk] = (_Float16)val;
        } else {
          Of[(size_t)m * 1024 + n] = val;
        }
      }
    }
  }
}

// ==========================================================================
// Flash attention: per block (b,h, 128 query rows). 8 waves, each wave owns
// 16 query rows. Key tiles of 64 positions: K staged via TDM DMA, V staged
// transposed manually. Online softmax, per-wave LDS round-trip converts P
// from C-layout to A-fragment layout for P @ V.
// Softmax scale (1/8) was folded into the Q projection.
// Reference mask semantics: score = -1e10 where mask[b, key] != 0.
// ==========================================================================
__global__ __launch_bounds__(256)
void attn_kernel(const _Float16* __restrict__ Qh, const _Float16* __restrict__ Kh,
                 const _Float16* __restrict__ Vh, const int* __restrict__ mask,
                 _Float16* __restrict__ Xc)
{
  constexpr int S = 2048;
  const int tid  = threadIdx.x;
  const int lane = tid & 31;
  const int wave = tid >> 5;
  const int half = lane >> 4;
  const int l15  = lane & 15;
  const int qb   = blockIdx.x;         // 0..15
  const int bh   = blockIdx.y;         // 0..31
  const int b    = bh >> 4, h = bh & 15;
  const int qRow0 = qb * 128 + wave * 16;

  __shared__ alignas(16) _Float16 smem[4096 + 4096 + 8192];   // 32 KB
  _Float16* Ks  = smem;                // [key(64)][dk(64)]
  _Float16* Vt  = smem + 4096;         // [dk(64)][key(64)]  (transposed)
  _Float16* Ps  = smem + 8192;         // per-wave 16x64 scratch
  _Float16* Psw = Ps + wave * 1024;

  // ---- stage this wave's Q tile (16x64, contiguous in Qh) and build frags
  {
    const _Float16* gq = Qh + ((size_t)bh * S + qRow0) * 64;
    for (int i = 0; i < 4; ++i) {
      const int off = i * 256 + lane * 8;
      *(v8h*)(Psw + off) = *(const v8h*)(gq + off);
    }
  }
  asm volatile("s_wait_dscnt 0" ::: "memory");
  v16h aq[2];
  for (int c = 0; c < 2; ++c)
    aq[c] = load_a_frag(Psw + l15 * 64 + c * 32 + 16 * half);

  float m_run[8], l_run[8];
  for (int r = 0; r < 8; ++r) { m_run[r] = -3.0e38f; l_run[r] = 0.0f; }
  v8f zero = {};
  v8f o[4];
  for (int dt = 0; dt < 4; ++dt) o[dt] = zero;

  for (int kt = 0; kt < S / 64; ++kt) {
    __syncthreads();   // all waves done reading previous Ks/Vt

    // prefetch next key tile into cache (global_prefetch_b8)
    if (kt + 1 < S / 64) {
      const _Float16* nk = Kh + ((size_t)bh * S + (kt + 1) * 64) * 64;
      const _Float16* nv = Vh + ((size_t)bh * S + (kt + 1) * 64) * 64;
      __builtin_prefetch(nk + tid * 16, 0, 1);
      __builtin_prefetch(nv + tid * 16, 0, 1);
    }

    // ---- stage K tile [key][dk] (contiguous 64x64 f16)
#if USE_TDM
    if (wave == 0)
      tdm_load_tile_f16((unsigned int)(uintptr_t)(void*)Ks,
                        Kh + ((size_t)bh * S + kt * 64) * 64, 64u, 64u, 64u);
#else
    {
      const _Float16* gk = Kh + ((size_t)bh * S + kt * 64) * 64;
      const int off = tid * 16;
      *(v8h*)(Ks + off)     = *(const v8h*)(gk + off);
      *(v8h*)(Ks + off + 8) = *(const v8h*)(gk + off + 8);
    }
#endif
    // ---- stage V tile transposed: Vt[dk][key] (TDM cannot transpose)
    {
      const _Float16* gv = Vh + ((size_t)bh * S + kt * 64) * 64;
      for (int i = tid; i < 4096; i += 256) {
        const int key = i >> 6, dk = i & 63;
        Vt[dk * 64 + key] = gv[i];
      }
    }
#if USE_TDM
    if (wave == 0)
      __builtin_amdgcn_s_wait_tensorcnt(0);
#endif
    __syncthreads();

    // ---- scores: 16 q-rows x 64 keys (4 tiles), K-dim 64 = 2 chunks
    v8f sc[4];
    for (int nt = 0; nt < 4; ++nt) {
      sc[nt] = zero;
      for (int c = 0; c < 2; ++c) {
        v16h bk = load_b_frag(Ks + (nt * 16 + l15) * 64 + c * 32 + 16 * half);
        sc[nt] = __builtin_amdgcn_wmma_f32_16x16x32_f16(
            false, aq[c], false, bk, (short)0, sc[nt], false, false);
      }
      const int mv = mask[(size_t)b * S + kt * 64 + nt * 16 + l15];
      if (mv != 0)
        for (int r = 0; r < 8; ++r) sc[nt][r] = -1.0e10f;
    }

    // ---- online softmax (rows live in 16-lane halves)
    float alpha[8], psum[8];
    for (int r = 0; r < 8; ++r) {
      float t = fmaxf(fmaxf(sc[0][r], sc[1][r]), fmaxf(sc[2][r], sc[3][r]));
      t = half_red_max(t);
      const float nm = fmaxf(m_run[r], t);
      alpha[r] = __expf(m_run[r] - nm);
      m_run[r] = nm;
      psum[r] = 0.0f;
    }
    for (int nt = 0; nt < 4; ++nt)
      for (int r = 0; r < 8; ++r) {
        const float pv = __expf(sc[nt][r] - m_run[r]);
        psum[r] += pv;
        Psw[(half * 8 + r) * 64 + nt * 16 + l15] = (_Float16)pv;
      }
    for (int r = 0; r < 8; ++r) {
      l_run[r] = l_run[r] * alpha[r] + half_red_sum(psum[r]);
      for (int dt = 0; dt < 4; ++dt) o[dt][r] *= alpha[r];
    }

    asm volatile("s_wait_dscnt 0" ::: "memory");

    // ---- O += P @ V
    for (int c = 0; c < 2; ++c) {
      v16h ap = load_a_frag(Psw + l15 * 64 + c * 32 + 16 * half);
      for (int dt = 0; dt < 4; ++dt) {
        v16h bv = load_b_frag(Vt + (dt * 16 + l15) * 64 + c * 32 + 16 * half);
        o[dt] = __builtin_amdgcn_wmma_f32_16x16x32_f16(
            false, ap, false, bv, (short)0, o[dt], false, false);
      }
    }
  }

  // ---- normalize + write merged-head layout [B,S,D] f16
  for (int dt = 0; dt < 4; ++dt)
    for (int r = 0; r < 8; ++r) {
      const int row = qRow0 + half * 8 + r;
      const float val = o[dt][r] / l_run[r];
      Xc[((size_t)b * S + row) * 1024 + h * 64 + dt * 16 + l15] = (_Float16)val;
    }
}

// ==========================================================================
extern "C" void kernel_launch(void* const* d_in, const int* in_sizes, int n_in,
                              void* d_out, int out_size, void* d_ws, size_t ws_size,
                              hipStream_t stream) {
  (void)in_sizes; (void)n_in; (void)out_size; (void)ws_size;
  const float* q    = (const float*)d_in[0];
  const float* k    = (const float*)d_in[1];
  const float* v    = (const float*)d_in[2];
  const int*   mask = (const int*)d_in[3];
  const float* Wq   = (const float*)d_in[4];
  const float* bq   = (const float*)d_in[5];
  const float* Wk   = (const float*)d_in[6];
  const float* bk   = (const float*)d_in[7];
  const float* Wv   = (const float*)d_in[8];
  const float* bv   = (const float*)d_in[9];
  const float* Wo   = (const float*)d_in[10];
  const float* bo   = (const float*)d_in[11];
  float* out = (float*)d_out;

  // workspace: 4 x (4096*1024) f16 buffers = 32 MB
  _Float16* ws = (_Float16*)d_ws;
  const size_t BUF = (size_t)4096 * 1024;
  _Float16* Qh = ws;
  _Float16* Kh = ws + BUF;
  _Float16* Vh = ws + 2 * BUF;
  _Float16* Xc = ws + 3 * BUF;

  const dim3 gemmGrid(8, 32);    // N/128, M/128
  const dim3 attnGrid(16, 32);   // S/128, B*H
  const dim3 blk(256);

  // projections (softmax scale 1/sqrt(64)=0.125 folded into Q)
  gemm_kernel<0><<<gemmGrid, blk, 0, stream>>>(q, Wq, bq, Qh, 0.125f);
  gemm_kernel<0><<<gemmGrid, blk, 0, stream>>>(k, Wk, bk, Kh, 1.0f);
  gemm_kernel<0><<<gemmGrid, blk, 0, stream>>>(v, Wv, bv, Vh, 1.0f);
  // attention
  attn_kernel<<<attnGrid, blk, 0, stream>>>(Qh, Kh, Vh, mask, Xc);
  // output projection (f32 out)
  gemm_kernel<1><<<gemmGrid, blk, 0, stream>>>(Xc, Wo, bo, out, 1.0f);
}
